// BoundaryHausdorffLoss_80745385165233
// MI455X (gfx1250) — matrix operand order repaired and checked by reading, not compiled, not run on hardware
//
#include <hip/hip_runtime.h>
#include <hip/hip_bf16.h>

// ---------------------------------------------------------------------------
// BoundaryHausdorffLoss on gfx1250.
// EDT min-plus passes lifted to the log semiring:
//   d2(i,w) = -T * ln( (E @ B @ E)[i,w] ),  E[i,j] = exp(-(i-j)^2 / T)
// Heavy work: 16 dense 512x512x512 f16 GEMMs on v_wmma_f32_16x16x32_f16.
// Operands are staged into LDS by the Tensor Data Mover (double buffered,
// s_wait_tensorcnt-paced); fragments come out of LDS as ds_load_b128;
// 4x4 register tiling gives 16 WMMAs per 32-wide k-chunk.
// ---------------------------------------------------------------------------

typedef __attribute__((ext_vector_type(16))) _Float16 v16h;
typedef __attribute__((ext_vector_type(8)))  float    v8f;
typedef __attribute__((ext_vector_type(4)))  unsigned int u32x4;
typedef __attribute__((ext_vector_type(4)))  int          i32x4;
typedef __attribute__((ext_vector_type(8)))  int          i32x8;

#define HW 512
#define NPIX (HW * HW)          // 2^18
#define PIX_SHIFT 18
#define NIMG 8                  // 0..3: pred boundary, 4..7: target boundary
#define SOFT_T 2.0f
#define SOFT_INV_T 0.5f
#define LOSS_EPS 1e-6f

// Workspace layout (bytes)
static constexpr size_t OFF_E    = 0;                                    // 512KB f16 Gaussian kernel (symmetric)
static constexpr size_t OFF_BND  = OFF_E    + (size_t)NPIX * 2;          // 4MB  f16 boundary maps (row-major)
static constexpr size_t OFF_BNDT = OFF_BND  + (size_t)NIMG * NPIX * 2;   // 4MB  f16 boundary maps (transposed)
static constexpr size_t OFF_S    = OFF_BNDT + (size_t)NIMG * NPIX * 2;   // 4MB  f16 stage-1 result
static constexpr size_t OFF_D    = OFF_S    + (size_t)NIMG * NPIX * 2;   // 8MB  f32 stage-2 result
static constexpr size_t OFF_ACC  = OFF_D    + (size_t)NIMG * NPIX * 4;   // 16B  accumulators

// --- TDM availability: probe-confirmed builtin, arity differs per toolchain ---
#ifdef __HIP_DEVICE_COMPILE__
#  ifdef __has_builtin
#    if __has_builtin(__builtin_amdgcn_tensor_load_to_lds) && \
        __has_builtin(__builtin_amdgcn_s_wait_tensorcnt)
#      define BHL_HAVE_TDM 1
#    endif
#  endif
#endif
#ifndef BHL_HAVE_TDM
#  define BHL_HAVE_TDM 0
#endif

// Wait for TENSORcnt <= n, then force the compiler to treat LDS as modified
// (the TDM intrinsic's LDS write is not modeled against our ds loads;
// __shared__ arrays are AS(3) globals so a full memory clobber pins the
// fragment loads after the wait and inside the loop).
#if BHL_HAVE_TDM
#  define BHL_WAIT_TENSOR(n)                                   \
      do {                                                     \
          __builtin_amdgcn_s_wait_tensorcnt((short)(n));       \
          __asm__ volatile("" ::: "memory");                   \
      } while (0)
#else
#  define BHL_WAIT_TENSOR(n) __asm__ volatile("" ::: "memory")
#endif

__global__ void bhl_init_acc(float* acc) {
    if (threadIdx.x < 4) acc[threadIdx.x] = 0.0f;
}

// E[i,j] = exp(-(i-j)^2 / T); symmetric, so it serves as both A and B^T operand.
__global__ void bhl_gauss(_Float16* __restrict__ E) {
    int gid = blockIdx.x * blockDim.x + threadIdx.x;
    if (gid >= NPIX) return;
    int i = gid >> 9, j = gid & (HW - 1);
    float d = (float)(i - j);
    E[gid] = (_Float16)__expf(-d * d * SOFT_INV_T);
}

// 3x3 boundary: dilation(OR over in-bounds) AND NOT erosion(AND with zero-pad).
// Writes both row-major and transposed copies (transposed feeds the GEMM B^T).
__global__ void bhl_boundary(const float* __restrict__ logits,
                             const int*   __restrict__ targets,
                             _Float16*    __restrict__ bnd,
                             _Float16*    __restrict__ bndT) {
    int gid = blockIdx.x * blockDim.x + threadIdx.x;
    if (gid >= NIMG * NPIX) return;
    int img = gid >> PIX_SHIFT;
    int pix = gid & (NPIX - 1);
    int y = pix >> 9, x = pix & (HW - 1);
    int b = img & 3;
    bool isPred = img < 4;
    const float* lg = logits + (size_t)b * NPIX;
    const int*   tg = targets + (size_t)b * NPIX;

    bool orAcc = false, andAcc = true;
#pragma unroll
    for (int dy = -1; dy <= 1; ++dy) {
#pragma unroll
        for (int dx = -1; dx <= 1; ++dx) {
            int yy = y + dy, xx = x + dx;
            bool inb = (yy >= 0) & (yy < HW) & (xx >= 0) & (xx < HW);
            bool m = false;
            if (inb) {
                int p = yy * HW + xx;
                m = isPred ? (lg[p] > 0.0f)   // sigmoid(z) > 0.5  <=>  z > 0
                           : (tg[p] != 0);
            }
            orAcc |= (inb && m);
            andAcc &= m;   // OOB counts as 0 -> erosion fails at border (zero-pad)
        }
    }
    _Float16 v = (orAcc && !andAcc) ? (_Float16)1.0f : (_Float16)0.0f;
    bnd[gid] = v;
    bndT[(size_t)img * NPIX + (size_t)x * HW + y] = v;
}

// C(64x64 per wave) = A(MxK) @ B, with B supplied transposed (BT is NxK
// row-major). K=512 in chunks of 32; chunks staged to LDS by the TDM
// (double buffered); fragments read as ds_load_b128 per ISA 7.12.2 layouts.
template <bool OUT_F32>
__global__ void __launch_bounds__(32)
bhl_gemm_wmma(const _Float16* __restrict__ A,  size_t aImgStride,
              const _Float16* __restrict__ BT, size_t bImgStride,
              void* __restrict__ Cout, size_t cImgStride) {
    // [buf:2][op:2] chunks of 64x32 f16 (4KB each) = 16KB LDS
    __shared__ _Float16 smem[4 * 2048];

    const int lane = threadIdx.x;      // wave32
    const int mr   = lane & 15;
    const int hf   = lane >> 4;
    const int img  = blockIdx.z;
    const int m0   = blockIdx.y * 64;
    const int n0   = blockIdx.x * 64;
    const _Float16* Abase = A  + (size_t)img * aImgStride + (size_t)m0 * HW;
    const _Float16* Bbase = BT + (size_t)img * bImgStride + (size_t)n0 * HW;

    union Frag { v16h h; uint4 q[2]; };
    v8f acc[4][4] = {};

#if BHL_HAVE_TDM
    // Loop-invariant descriptor group 1 (ISA §8.4):
    //   data_size=2B, tensor_dim0=tensor_dim1=512, tile=[32 x 64], stride0=512.
    i32x8 g1;
    g1[0] = 0x00010000;
    g1[1] = (int)((512u & 0xFFFFu) << 16);                               // dim0.lo @ bits 79:48
    g1[2] = (int)(((512u >> 16) & 0xFFFFu) | ((512u & 0xFFFFu) << 16));  // dim0.hi | dim1.lo
    g1[3] = (int)(((512u >> 16) & 0xFFFFu) | (32u << 16));               // dim1.hi | tile_dim0
    g1[4] = 64;                                                          // tile_dim1 | tile_dim2=0
    g1[5] = 512;                                                         // stride0.lo
    g1[6] = 0;
    g1[7] = 0;
    const i32x4 gz4 = {0, 0, 0, 0};
#if __clang_major__ >= 23
    const i32x8 gz8 = {0, 0, 0, 0, 0, 0, 0, 0};
#endif
    auto tdm_load = [&](unsigned ldsOff, const _Float16* gsrc) {
        unsigned long long ga = (unsigned long long)(uintptr_t)gsrc;
        u32x4 g0;
        g0[0] = 1u;                                                   // count=1 user desc
        g0[1] = ldsOff;                                               // lds_addr (bytes)
        g0[2] = (unsigned)ga;                                         // addr[31:0]
        g0[3] = (unsigned)((ga >> 32) & 0x01FFFFFFu) | 0x80000000u;   // addr[56:32] | type=2
#if __clang_major__ >= 23
        __builtin_amdgcn_tensor_load_to_lds(g0, g1, gz4, gz4, gz8, 0);
#else
        __builtin_amdgcn_tensor_load_to_lds(g0, g1, gz4, gz4, 0);
#endif
    };
#endif

    auto load_chunk = [&](int buf, int k0) {
#if BHL_HAVE_TDM
        tdm_load((unsigned)((buf * 2 + 0) * 4096), Abase + k0);
        tdm_load((unsigned)((buf * 2 + 1) * 4096), Bbase + k0);
#else
        uint4* da = (uint4*)&smem[(buf * 2 + 0) * 2048];
        uint4* db = (uint4*)&smem[(buf * 2 + 1) * 2048];
#pragma unroll
        for (int i = 0; i < 8; ++i) {
            int t = lane + 32 * i;           // 256 16-byte chunks per tile
            int row = t >> 2, part = t & 3;  // 4 x uint4 per 32-half row
            da[t] = ((const uint4*)(Abase + k0 + (size_t)row * HW))[part];
            db[t] = ((const uint4*)(Bbase + k0 + (size_t)row * HW))[part];
        }
#endif
    };

    auto compute = [&](int buf) {
        const _Float16* ab = &smem[(buf * 2 + 0) * 2048];
        const _Float16* bb = &smem[(buf * 2 + 1) * 2048];
        Frag fa[4];
#pragma unroll
        for (int m = 0; m < 4; ++m) {
            // A 16x32 f16 layout: lane holds row; dwords [4h..4h+3], [8+4h..11+4h]
            const uint4* p = (const uint4*)(ab + (m * 16 + mr) * 32);
            fa[m].q[0] = p[hf];
            fa[m].q[1] = p[2 + hf];
        }
#pragma unroll
        for (int n = 0; n < 4; ++n) {
            // B 32x16 f16 layout via BT rows: lane holds column; dwords [4h..4h+3], [8+4h..11+4h]
            Frag fb;
            const uint4* p = (const uint4*)(bb + (n * 16 + mr) * 32);
            fb.q[0] = p[hf];
            fb.q[1] = p[2 + hf];
#pragma unroll
            for (int m = 0; m < 4; ++m) {
                acc[m][n] = __builtin_amdgcn_wmma_f32_16x16x32_f16(
                    false, fa[m].h, false, fb.h, (short)0, acc[m][n],
                    false, false);
            }
        }
    };

    int cur = 0;
    load_chunk(cur, 0);                       // prime the pipeline
#pragma unroll 1
    for (int k0 = 0; k0 < HW - 32; k0 += 32) {
        load_chunk(cur ^ 1, k0 + 32);         // prefetch next chunk (2 TDM ops)
        BHL_WAIT_TENSOR(2);                   // current chunk landed in LDS
        compute(cur);
        cur ^= 1;
    }
    BHL_WAIT_TENSOR(0);
    compute(cur);

    // C/D layout: lane holds column n = lane&15; VGPR r holds row r + 8*(lane>>4)
    const size_t cBase = (size_t)img * cImgStride;
#pragma unroll
    for (int m = 0; m < 4; ++m) {
#pragma unroll
        for (int n = 0; n < 4; ++n) {
#pragma unroll
            for (int r = 0; r < 8; ++r) {
                size_t idx = cBase + (size_t)(m0 + m * 16 + r + 8 * hf) * HW
                           + (n0 + n * 16 + mr);
                if constexpr (OUT_F32) ((float*)Cout)[idx]     = acc[m][n][r];
                else                   ((_Float16*)Cout)[idx]  = (_Float16)acc[m][n][r];
            }
        }
    }
}

// d2 = max(0, -T*ln(D)); accumulate pd = sum d2_pred*tb, td = sum d2_tgt*pb,
// plus the boundary-pixel counts.
__global__ void bhl_reduce(const float*    __restrict__ D,
                           const _Float16* __restrict__ bnd,
                           float*          __restrict__ acc) {
    __shared__ float sm[4][256];
    int gid = blockIdx.x * blockDim.x + threadIdx.x;
    float pd = 0.f, td = 0.f, tb = 0.f, pb = 0.f;
    if (gid < 4 * NPIX) {
        int bimg = gid >> PIX_SHIFT;
        int pix  = gid & (NPIX - 1);
        float bp = (float)bnd[(size_t)bimg * NPIX + pix];
        float bt = (float)bnd[(size_t)(bimg + 4) * NPIX + pix];
        float Dp = D[(size_t)bimg * NPIX + pix];
        float Dt = D[(size_t)(bimg + 4) * NPIX + pix];
        float d2p = (Dp > 1e-30f) ? fmaxf(0.0f, -SOFT_T * __logf(Dp)) : 1e8f;
        float d2t = (Dt > 1e-30f) ? fmaxf(0.0f, -SOFT_T * __logf(Dt)) : 1e8f;
        pd = d2p * bt;  td = d2t * bp;  tb = bt;  pb = bp;
    }
    int t = threadIdx.x;
    sm[0][t] = pd; sm[1][t] = td; sm[2][t] = tb; sm[3][t] = pb;
    __syncthreads();
    for (int s = 128; s > 0; s >>= 1) {
        if (t < s) {
#pragma unroll
            for (int q = 0; q < 4; ++q) sm[q][t] += sm[q][t + s];
        }
        __syncthreads();
    }
    if (t == 0) {
#pragma unroll
        for (int q = 0; q < 4; ++q) atomicAdd(&acc[q], sm[q][0]);
    }
}

__global__ void bhl_finalize(const float* __restrict__ acc, float* __restrict__ out) {
    if (threadIdx.x == 0 && blockIdx.x == 0) {
        float pred_loss   = acc[0] / (acc[2] + LOSS_EPS);  // / (sum target boundary + eps)
        float target_loss = acc[1] / (acc[3] + LOSS_EPS);  // / (sum pred boundary + eps)
        out[0] = 0.5f * (pred_loss + target_loss);
    }
}

extern "C" void kernel_launch(void* const* d_in, const int* in_sizes, int n_in,
                              void* d_out, int out_size, void* d_ws, size_t ws_size,
                              hipStream_t stream) {
    const float* logits  = (const float*)d_in[0];
    const int*   targets = (const int*)d_in[1];
    float* out = (float*)d_out;
    char*  ws  = (char*)d_ws;

    _Float16* E    = (_Float16*)(ws + OFF_E);
    _Float16* bnd  = (_Float16*)(ws + OFF_BND);
    _Float16* bndT = (_Float16*)(ws + OFF_BNDT);
    _Float16* S    = (_Float16*)(ws + OFF_S);
    float*    D    = (float*)(ws + OFF_D);
    float*    acc  = (float*)(ws + OFF_ACC);

    bhl_init_acc<<<1, 32, 0, stream>>>(acc);
    bhl_gauss<<<NPIX / 256, 256, 0, stream>>>(E);
    bhl_boundary<<<(NIMG * NPIX) / 256, 256, 0, stream>>>(logits, targets, bnd, bndT);

    dim3 gtiles(HW / 64, HW / 64, NIMG);
    // Stage 1 (soft column pass): S = E @ B      (B^T = transposed boundary map)
    bhl_gemm_wmma<false><<<gtiles, 32, 0, stream>>>(E, 0, bndT, (size_t)NPIX,
                                                    S, (size_t)NPIX);
    // Stage 2 (soft row pass):    D = S @ E      (E symmetric => E^T = E)
    bhl_gemm_wmma<true><<<gtiles, 32, 0, stream>>>(S, (size_t)NPIX, E, 0,
                                                   D, (size_t)NPIX);

    bhl_reduce<<<(4 * NPIX) / 256, 256, 0, stream>>>(D, bnd, acc);
    bhl_finalize<<<1, 64, 0, stream>>>(acc, out);
}
